// SimpleSelfAttentionModelV0_3951369913024
// MI455X (gfx1250) — matrix operand order repaired and checked by reading, not compiled, not run on hardware
//
#include <hip/hip_runtime.h>
#include <hip/hip_bf16.h>
#include <math.h>

// Problem constants (match reference)
#define B_    16
#define N_    2048
#define E_    512
#define OUT_  8

#define KROW  520           // padded LDS row stride (shorts): 1040B = 65*16B,
                            // shifts banks by 4 per row -> conflict-free b128 reads
#define LOG2E 1.44269504088896340736f

// CDNA5 WMMA fragment types (wave32): 16 bf16 = 8 VGPRs, 8 f32 = 8 VGPRs
typedef __attribute__((ext_vector_type(16))) __bf16 v16bf;
typedef __attribute__((ext_vector_type(8)))  float  v8f;

union BF16Frag { v16bf v; uint4 q[2]; };

__device__ __forceinline__ unsigned short f2bf(float f) {
    union { float f; unsigned u; } a; a.f = f;
    unsigned r = a.u + 0x7FFFu + ((a.u >> 16) & 1u);   // round-to-nearest-even
    return (unsigned short)(r >> 16);
}

// ---------------------------------------------------------------------------
// 1) Embedding gather + fp32 -> bf16 cast, padding_idx==0 forced to zero.
// ---------------------------------------------------------------------------
__global__ __launch_bounds__(64) void gather_cast(
    const int* __restrict__ x, const float* __restrict__ table,
    unsigned short* __restrict__ embbf) {
    int row = blockIdx.x;                 // 0 .. B*N-1
    int idx = x[row];
    const float* src = table + (long)idx * E_;
    int e0 = threadIdx.x * 8;
    float4 f0, f1;
    if (idx == 0) { f0 = make_float4(0.f,0.f,0.f,0.f); f1 = f0; }
    else { f0 = *(const float4*)(src + e0); f1 = *(const float4*)(src + e0 + 4); }
    unsigned r0 = f2bf(f0.x) | ((unsigned)f2bf(f0.y) << 16);
    unsigned r1 = f2bf(f0.z) | ((unsigned)f2bf(f0.w) << 16);
    unsigned r2 = f2bf(f1.x) | ((unsigned)f2bf(f1.y) << 16);
    unsigned r3 = f2bf(f1.z) | ((unsigned)f2bf(f1.w) << 16);
    uint4 o; o.x = r0; o.y = r1; o.z = r2; o.w = r3;
    *(uint4*)(embbf + (long)row * E_ + e0) = o;       // 16B coalesced store
}

// ---------------------------------------------------------------------------
// 2) LDS-tiled transpose: vt[b][e][n] = embbf[b][n][e]  (bf16)
// ---------------------------------------------------------------------------
__global__ __launch_bounds__(256) void transpose_bf(
    const unsigned short* __restrict__ embbf, unsigned short* __restrict__ vt) {
    __shared__ unsigned short tile[64][65];           // +1 pad vs bank conflicts
    int b  = blockIdx.z;
    int n0 = blockIdx.x * 64;
    int e0 = blockIdx.y * 64;
    int tx = threadIdx.x & 63, ty = threadIdx.x >> 6; // 64 x 4
    const unsigned short* src = embbf + (long)b * N_ * E_;
#pragma unroll
    for (int r = 0; r < 16; ++r) {
        int nl = ty + r * 4;
        tile[nl][tx] = src[(long)(n0 + nl) * E_ + e0 + tx];
    }
    __syncthreads();
    unsigned short* dst = vt + (long)b * E_ * N_;
#pragma unroll
    for (int r = 0; r < 16; ++r) {
        int el = ty + r * 4;
        dst[(long)(e0 + el) * N_ + n0 + tx] = tile[tx][el];
    }
}

// ---------------------------------------------------------------------------
// 3) Pooled-max buffer init (monotonic uint keys; 0 < key(-inf))
// ---------------------------------------------------------------------------
__global__ void init_pooled(unsigned* pooled) {
    pooled[blockIdx.x * 256 + threadIdx.x] = 0u;
}

// ---------------------------------------------------------------------------
// 4) Flash attention. Block = 4 waves = 4 query tiles of the SAME batch,
//    sharing one key stream. The 32x512 bf16 K-tile is staged once per block
//    into LDS (padded rows), cutting QK^T L2 traffic 4x; all four waves read
//    B-fragments via ds_load_b128. P*V B-fragments stream from the global
//    V^T copy (L2-resident, long clause offsets pipeline with WMMA).
//    ctx accumulator: 16x512 f32 = 32 v8f C-frags (256 VGPRs, wave32-legal).
//    Softmax runs in base-2 domain (scores pre-scaled by log2 e) so every
//    exponential is a bare v_exp_f32.
// ---------------------------------------------------------------------------
__global__ __launch_bounds__(128, 1) void flash_attn(
    const unsigned short* __restrict__ embbf,
    const unsigned short* __restrict__ vt,
    unsigned* __restrict__ pooled) {
    __shared__ __align__(16) unsigned short Kbuf[32 * KROW];       // 33,280 B
    __shared__ __align__(16) unsigned short p_lds_all[4][16 * 32]; // 4 KB

    const int wave = threadIdx.x >> 5;
    const int lane = threadIdx.x & 31;
    const int tile = blockIdx.x * 4 + wave;           // 0 .. 2047
    const int b    = tile >> 7;                       // 128 q-tiles per batch
    const int q0   = (tile & 127) << 4;
    unsigned short* p_lds = p_lds_all[wave];

    const unsigned short* Qb  = embbf + (long)b * N_ * E_;
    const unsigned short* Vtb = vt    + (long)b * E_ * N_;

    const int lr = lane & 15;
    const int hi = lane >> 4;                         // 0|1: lane half

    // Staging assignment: 4 threads per key row, 16 uint4 (256B) each
    const int srow = threadIdx.x >> 2;                // 0..31
    const int scol = (threadIdx.x & 3) * 16;          // uint4 index in row

    // --- Resident Q A-fragments (16x32 bf16 each, contiguous k = 16*hi + s per lane half) ---
    BF16Frag qf[16];
    {
        const unsigned short* qrow = Qb + (long)(q0 + lr) * E_;
#pragma unroll
        for (int kk = 0; kk < 16; ++kk) {
            int c = kk * 32 + hi * 16;
            qf[kk].q[0] = *(const uint4*)(qrow + c);       // K 0..7 | 16..23
            qf[kk].q[1] = *(const uint4*)(qrow + c + 8);   // K 8..15 | 24..31
        }
    }

    v8f acc[32];
#pragma unroll
    for (int f = 0; f < 32; ++f) acc[f] = v8f{};
    float m[8], l[8];
#pragma unroll
    for (int v = 0; v < 8; ++v) { m[v] = -INFINITY; l[v] = 0.f; }

    for (int kb = 0; kb < N_; kb += 32) {
        // ---- stage K tile (32 x 512 bf16) into padded LDS, all 4 waves ----
        __syncthreads();   // previous tile's LDS reads are done
        {
            const uint4* gsrc = (const uint4*)(Qb + (long)kb * E_)
                              + srow * (E_ / 8) + scol;
            uint4* ldst = (uint4*)(Kbuf + srow * KROW) + scol;
#pragma unroll
            for (int i = 0; i < 16; ++i) ldst[i] = gsrc[i];
        }
        __syncthreads();   // tile visible to all waves

        // ---- S(16x32) = Q * K^T : B-frag rows contiguous in LDS K-tile ----
        v8f Sa = v8f{}, Sb = v8f{};
        const unsigned short* krow0 = Kbuf + lr * KROW + hi * 16;
        const unsigned short* krow1 = krow0 + 16 * KROW;   // keys 16..31
#pragma unroll
        for (int kk = 0; kk < 16; ++kk) {
            BF16Frag b0, b1;
            b0.q[0] = *(const uint4*)(krow0 + kk * 32);
            b0.q[1] = *(const uint4*)(krow0 + kk * 32 + 8);
            b1.q[0] = *(const uint4*)(krow1 + kk * 32);
            b1.q[1] = *(const uint4*)(krow1 + kk * 32 + 8);
            Sa = __builtin_amdgcn_wmma_f32_16x16x32_bf16(false, qf[kk].v, false, b0.v,
                                                         (short)0, Sa, false, false);
            Sb = __builtin_amdgcn_wmma_f32_16x16x32_bf16(false, qf[kk].v, false, b1.v,
                                                         (short)0, Sb, false, false);
        }

        // ---- online softmax in base-2 (C-layout: VGPR v = row v+8*hi, col lr) ----
        float sc[8];
        bool changed = false;
#pragma unroll
        for (int v = 0; v < 8; ++v) {
            Sa[v] *= LOG2E;            // fold exp -> exp2 (softmax invariant)
            Sb[v] *= LOG2E;
            float t = fmaxf(Sa[v], Sb[v]);
            t = fmaxf(t, __shfl_xor(t, 1));
            t = fmaxf(t, __shfl_xor(t, 2));
            t = fmaxf(t, __shfl_xor(t, 4));
            t = fmaxf(t, __shfl_xor(t, 8));
            float nm = fmaxf(m[v], t);
            sc[v] = exp2f(m[v] - nm);  // bare v_exp_f32
            changed |= (nm != m[v]);
            m[v] = nm;
        }
#pragma unroll
        for (int v = 0; v < 8; ++v) {
            float pa = exp2f(Sa[v] - m[v]);
            float pb = exp2f(Sb[v] - m[v]);
            int r = v + hi * 8;
            p_lds[r * 32 + lr]      = f2bf(pa);           // C-layout -> [row][key]
            p_lds[r * 32 + 16 + lr] = f2bf(pb);
            float s = pa + pb;
            s += __shfl_xor(s, 1);
            s += __shfl_xor(s, 2);
            s += __shfl_xor(s, 4);
            s += __shfl_xor(s, 8);
            l[v] = l[v] * sc[v] + s;
        }
        // rescale accumulator only while running max still moves (diag-dominant)
        if (__any(changed)) {
#pragma unroll
            for (int f = 0; f < 32; ++f)
#pragma unroll
                for (int v = 0; v < 8; ++v) acc[f][v] *= sc[v];
        }

        __builtin_amdgcn_wave_barrier();  // keep compiler from hoisting P reads
        // ---- reload P as an A-fragment (A-layout) ----
        BF16Frag pf;
        {
            const unsigned short* prow = p_lds + lr * 32 + hi * 16;
            pf.q[0] = *(const uint4*)(prow);
            pf.q[1] = *(const uint4*)(prow + 8);
        }
        // ---- ctx += P * V : B-frag (k,n) = Vt[d=n][key=k] contiguous per lane ----
        const unsigned short* vrow = Vtb + (long)lr * N_ + kb + hi * 16;
#pragma unroll
        for (int f = 0; f < 32; ++f) {
            BF16Frag bv;
            const unsigned short* p = vrow + (long)f * 16 * N_;
            bv.q[0] = *(const uint4*)(p);
            bv.q[1] = *(const uint4*)(p + 8);
            acc[f] = __builtin_amdgcn_wmma_f32_16x16x32_bf16(false, pf.v, false, bv.v,
                                                             (short)0, acc[f], false, false);
        }
    }

    // ---- normalize, tile-local max over 16 rows, global atomic max ----
    float rinv[8];
#pragma unroll
    for (int v = 0; v < 8; ++v) rinv[v] = 1.f / l[v];
#pragma unroll
    for (int f = 0; f < 32; ++f) {
        float mx = -INFINITY;
#pragma unroll
        for (int v = 0; v < 8; ++v) mx = fmaxf(mx, acc[f][v] * rinv[v]);
        mx = fmaxf(mx, __shfl_xor(mx, 16));           // combine rows 0-7 / 8-15
        if (hi == 0) {
            union { float f; unsigned u; } a; a.f = mx;
            unsigned key = (a.u >> 31) ? ~a.u : (a.u | 0x80000000u); // order-preserving
            atomicMax(pooled + b * E_ + f * 16 + lr, key);
        }
    }
}

// ---------------------------------------------------------------------------
// 5) Epilogue: out[b][o] = pooled[b] . W[o] + bias[o]   (tiny: 128 dots of 512)
// ---------------------------------------------------------------------------
__global__ __launch_bounds__(128) void final_linear(
    const unsigned* __restrict__ pooledKey, const float* __restrict__ W,
    const float* __restrict__ bias, float* __restrict__ out) {
    int t = threadIdx.x;           // 0..127
    int b = t >> 3, o = t & 7;
    float s = 0.f;
    for (int e = 0; e < E_; ++e) {
        unsigned k = pooledKey[b * E_ + e];
        unsigned bits = (k >> 31) ? (k ^ 0x80000000u) : ~k;  // decode key -> f32
        s += __uint_as_float(bits) * W[o * E_ + e];
    }
    out[b * OUT_ + o] = s + bias[o];
}

// ---------------------------------------------------------------------------
// Launch: needs ws >= 64MB + 32KB (bf16 emb, bf16 emb^T, pooled keys)
// ---------------------------------------------------------------------------
extern "C" void kernel_launch(void* const* d_in, const int* in_sizes, int n_in,
                              void* d_out, int out_size, void* d_ws, size_t ws_size,
                              hipStream_t stream) {
    const int*   x     = (const int*)d_in[0];
    const float* table = (const float*)d_in[1];
    const float* W     = (const float*)d_in[2];
    const float* bias  = (const float*)d_in[3];
    float*       out   = (float*)d_out;

    unsigned short* embbf = (unsigned short*)d_ws;                       // 32 MB
    unsigned short* vtb   = embbf + (size_t)B_ * N_ * E_;                // 32 MB
    unsigned*       pool  = (unsigned*)(vtb + (size_t)B_ * N_ * E_);     // 32 KB

    gather_cast<<<B_ * N_, 64, 0, stream>>>(x, table, embbf);
    transpose_bf<<<dim3(N_ / 64, E_ / 64, B_), 256, 0, stream>>>(embbf, vtb);
    init_pooled<<<(B_ * E_) / 256, 256, 0, stream>>>(pool);
    flash_attn<<<(B_ * (N_ / 16)) / 4, 128, 0, stream>>>(embbf, vtb, pool);
    final_linear<<<1, 128, 0, stream>>>(pool, W, bias, out);
}
